// LSTMEncoder_29832842838366
// MI455X (gfx1250) — matrix operand order repaired and checked by reading, not compiled, not run on hardware
//
#include <hip/hip_runtime.h>

typedef _Float16 v4h  __attribute__((ext_vector_type(4)));
typedef _Float16 v8h  __attribute__((ext_vector_type(8)));
typedef _Float16 v16h __attribute__((ext_vector_type(16)));
typedef float    v8f  __attribute__((ext_vector_type(8)));

#define T_SEQ 1024
#define KDIM  128   // inner dim of every input transform (D = H = 128)

// ---------------------------------------------------------------- helpers

__device__ __forceinline__ float sigmoidf_(float x) {
    return 1.0f / (1.0f + __expf(-x));
}
__device__ __forceinline__ float tanhf_(float x) {
    return 2.0f * sigmoidf_(2.0f * x) - 1.0f;
}

// deterministic per-(gate,batch,d) dropout mask, time-invariant (input dropout
// with keep = 0.4 -> value 1/0.4 = 2.5 or 0).
__device__ __forceinline__ float dropmask(unsigned salt, int g, int b, int d) {
    unsigned h = salt;
    h ^= (unsigned)g * 0x9E3779B9u;
    h ^= (unsigned)b * 0x85EBCA6Bu;
    h ^= (unsigned)d * 0xC2B2AE35u;
    h ^= h >> 16; h *= 0x7FEB352Du;
    h ^= h >> 15; h *= 0x846CA68Bu;
    h ^= h >> 16;
    float u = (float)(h >> 8) * (1.0f / 16777216.0f);
    return (u < 0.4f) ? 2.5f : 0.0f;
}

__device__ __forceinline__ v16h cat8(v8h lo, v8h hi) {
    v16h r;
#pragma unroll
    for (int i = 0; i < 8; ++i) { r[i] = lo[i]; r[i + 8] = hi[i]; }
    return r;
}

__device__ __forceinline__ v8f zero8() {
    v8f r;
#pragma unroll
    for (int i = 0; i < 8; ++i) r[i] = 0.0f;
    return r;
}

// A fragment (16xK32 f16) from a row-major LDS tile, 'stride' halves per row.
// ISA 7.12.2: lane<16 -> M=lane, halves 0..7 = K 0..7, halves 8..15 = K 16..23
//             lane>=16 -> M=lane-16, K 8..15 and K 24..31.
__device__ __forceinline__ v16h afrag(const _Float16* tile, int stride, int lane) {
    int m = lane & 15, sel = lane >> 4;
    v8h lo = *(const v8h*)(tile + m * stride + sel * 8);
    v8h hi = *(const v8h*)(tile + m * stride + 16 + sel * 8);
    return cat8(lo, hi);
}

// B fragment (K32 x 16 f16) from a column-major LDS tile ([col][stride] halves).
// lanes 0-15: col=lane, K = kb+0..7 and kb+16..23 ; lanes 16-31: col=lane-16, K = kb+8..15 and kb+24..31.
__device__ __forceinline__ v16h bfrag(const _Float16* tile, int stride, int lane,
                                      int cb, int kb) {
    int n = lane & 15, sel = lane >> 4;
    const _Float16* p = tile + (size_t)(cb + n) * stride + kb + sel * 8;
    return cat8(*(const v8h*)p, *(const v8h*)(p + 16));
}

// ---------------------------------------------------------- input transform
// Z[M,4U] = (A * mask_gate) @ Wk + bias  ; A is [M,128] f32, time-invariant mask.
// Block = 256 threads (8 waves), tile = 128 rows x 64 cols, K = 128.
template <int U>
__global__ __launch_bounds__(256) void lstm_in_gemm(
    const float* __restrict__ A, const float* __restrict__ Wk,
    const float* __restrict__ bias, float* __restrict__ Z, unsigned salt) {
    constexpr int N4 = 4 * U;
    constexpr int AP = 40;  // Asub row pitch in halves (80B, 16B aligned)
    __shared__ _Float16 Asub[128 * AP];
    __shared__ _Float16 WkT[64 * KDIM];  // [col][k] f16

    const int tid = threadIdx.x;
    const int lane = tid & 31, wave = tid >> 5;
    const int rowbase = blockIdx.x * 128;
    const int colbase = blockIdx.y * 64;
    const int gate = colbase / U;

    // transpose+convert the Wk column panel once
    for (int idx = tid; idx < 128 * 64; idx += 256) {
        int k = idx >> 6, c = idx & 63;
        WkT[c * KDIM + k] = (_Float16)Wk[(size_t)k * N4 + colbase + c];
    }

    v8f acc[4];
#pragma unroll
    for (int tc = 0; tc < 4; ++tc) acc[tc] = zero8();

    __syncthreads();

    for (int kb = 0; kb < KDIM; kb += 32) {
        // stage masked A tile (128x32) as f16
#pragma unroll
        for (int q = 0; q < 4; ++q) {
            int i = q * 256 + tid;          // 1024 float4 loads
            int r = i >> 3, kk = (i & 7) * 4;
            int grow = rowbase + r;
            int b = grow >> 10;             // batch index (T = 1024)
            float4 v = *(const float4*)(A + (size_t)grow * KDIM + kb + kk);
            v4h hv;
            hv[0] = (_Float16)(v.x * dropmask(salt, gate, b, kb + kk + 0));
            hv[1] = (_Float16)(v.y * dropmask(salt, gate, b, kb + kk + 1));
            hv[2] = (_Float16)(v.z * dropmask(salt, gate, b, kb + kk + 2));
            hv[3] = (_Float16)(v.w * dropmask(salt, gate, b, kb + kk + 3));
            *(v4h*)(Asub + r * AP + kk) = hv;
        }
        __syncthreads();

        v16h a = afrag(Asub + (wave * 16) * AP, AP, lane);
        v16h bfv[4];
#pragma unroll
        for (int tc = 0; tc < 4; ++tc)
            bfv[tc] = bfrag(WkT, KDIM, lane, tc * 16, kb);
#pragma unroll
        for (int tc = 0; tc < 4; ++tc)
            acc[tc] = __builtin_amdgcn_wmma_f32_16x16x32_f16(
                false, a, false, bfv[tc], (short)0, acc[tc], false, false);
        __syncthreads();
    }

    // epilogue: + bias, store f32
    int n = lane & 15, msel = (lane >> 4) * 8;
#pragma unroll
    for (int tc = 0; tc < 4; ++tc) {
        float bv = bias[colbase + tc * 16 + n];
#pragma unroll
        for (int r = 0; r < 8; ++r) {
            int grow = rowbase + wave * 16 + msel + r;
            Z[(size_t)grow * N4 + colbase + tc * 16 + n] = acc[tc][r] + bv;
        }
    }
}

// --------------------------------------------------------------- scan layer
// Each block owns 16 batch rows; loops t = 0..1023. Wr fragments are hoisted
// into registers (constant over t); h lives in LDS as the WMMA A operand;
// c lives in VGPRs; Z_t is register-prefetched one step ahead (ping-pong).
template <int U, bool LAST>
__global__ __launch_bounds__(256) void lstm_scan_k(
    const float* __restrict__ Z,    // [B*T, 4U] precomputed input part (+bias)
    const float* __restrict__ Wr,   // [U, 4U]
    float* __restrict__ Hseq,       // [B*T, U]  (unused when LAST)
    float* __restrict__ outLast) {  // [B, U]    (LAST only)
    constexpr int N4 = 4 * U;
    constexpr int CPW = N4 / 8;        // columns per wave
    constexpr int TPW = CPW / 16;      // 16x16 tiles per wave
    constexpr int KI = U / 32;         // K-iterations
    constexpr int NE = 16 * U / 256;   // elementwise items per thread (contig u)
    constexpr int NV = NE / 4;         // float4s per gate per thread

    __shared__ _Float16 WrT[N4 * U];   // [col][k] f16 (used only to build bW)
    __shared__ _Float16 hbuf[16 * U];  // h_{t-1}, f16
    __shared__ float zbuf[16 * N4];    // recurrent part of z

    const int tid = threadIdx.x;
    const int lane = tid & 31, wave = tid >> 5;
    const int rowbase = blockIdx.x * 16;  // batch rows
    const int wcol = wave * CPW;
    const int mrow = (tid * NE) / U;      // this thread's batch row (0..15)
    const int u0 = (tid * NE) & (U - 1);  // first owned u (contiguous NE)

    for (int idx = tid; idx < N4 * U; idx += 256) {
        int k = idx / N4, c = idx & (N4 - 1);
        WrT[c * U + k] = (_Float16)Wr[idx];
    }
    for (int idx = tid; idx < 16 * U; idx += 256) hbuf[idx] = (_Float16)0.0f;
    __syncthreads();

    // Hoist all Wr B-fragments into registers: constant across all 1024 steps.
    v16h bW[KI * TPW];
#pragma unroll
    for (int ki = 0; ki < KI; ++ki)
#pragma unroll
        for (int tc = 0; tc < TPW; ++tc)
            bW[ki * TPW + tc] = bfrag(WrT, U, lane, wcol + tc * 16, ki * 32);

    float c_st[NE], h_st[NE];
#pragma unroll
    for (int e = 0; e < NE; ++e) { c_st[e] = 0.0f; h_st[e] = 0.0f; }

    const float* Zt0 = Z + (size_t)(rowbase + mrow) * T_SEQ * N4 + u0;

    auto prefetchZ = [&](int t, float4 (&zr)[4 * NV]) {
        if (t >= T_SEQ) return;
        const float* p = Zt0 + (size_t)t * N4;
#pragma unroll
        for (int g = 0; g < 4; ++g)
#pragma unroll
            for (int q = 0; q < NV; ++q)
                zr[g * NV + q] = *(const float4*)(p + g * U + q * 4);
    };

    auto step = [&](int t, float4 (&zc)[4 * NV], float4 (&zn)[4 * NV]) {
        // ---- recurrent GEMM: z_rec = h_{t-1} @ Wr
        v8f acc[TPW];
#pragma unroll
        for (int tc = 0; tc < TPW; ++tc) acc[tc] = zero8();
#pragma unroll
        for (int ki = 0; ki < KI; ++ki) {
            v16h a = afrag(hbuf + ki * 32, U, lane);
#pragma unroll
            for (int tc = 0; tc < TPW; ++tc)
                acc[tc] = __builtin_amdgcn_wmma_f32_16x16x32_f16(
                    false, a, false, bW[ki * TPW + tc], (short)0, acc[tc],
                    false, false);
        }
        {
            int n = lane & 15, msel = (lane >> 4) * 8;
#pragma unroll
            for (int tc = 0; tc < TPW; ++tc)
#pragma unroll
                for (int r = 0; r < 8; ++r)
                    zbuf[(msel + r) * N4 + wcol + tc * 16 + n] = acc[tc][r];
        }
        prefetchZ(t + 1, zn);  // hide global latency behind barrier+elementwise+GEMM
        __syncthreads();       // zbuf ready; all hbuf reads done

        // ---- elementwise gate math on NE contiguous u's
        float zg_[4][NE];
#pragma unroll
        for (int g = 0; g < 4; ++g)
#pragma unroll
            for (int q = 0; q < NV; ++q) {
                float4 l = *(const float4*)(&zbuf[mrow * N4 + g * U + u0 + q * 4]);
                float4 c4 = zc[g * NV + q];
                zg_[g][q * 4 + 0] = l.x + c4.x;
                zg_[g][q * 4 + 1] = l.y + c4.y;
                zg_[g][q * 4 + 2] = l.z + c4.z;
                zg_[g][q * 4 + 3] = l.w + c4.w;
            }
#pragma unroll
        for (int e = 0; e < NE; ++e) {
            float ig = sigmoidf_(zg_[0][e]), fg = sigmoidf_(zg_[1][e]);
            float gg = tanhf_(zg_[2][e]), og = sigmoidf_(zg_[3][e]);
            float c = fg * c_st[e] + ig * gg;
            float h = og * tanhf_(c);
            c_st[e] = c;
            h_st[e] = h;
        }
        if constexpr (NE == 8) {
            v8h hv;
#pragma unroll
            for (int e = 0; e < 8; ++e) hv[e] = (_Float16)h_st[e];
            *(v8h*)(hbuf + mrow * U + u0) = hv;
        } else {
            v4h hv;
#pragma unroll
            for (int e = 0; e < 4; ++e) hv[e] = (_Float16)h_st[e];
            *(v4h*)(hbuf + mrow * U + u0) = hv;
        }
        if (!LAST) {
            float* hp = Hseq + ((size_t)(rowbase + mrow) * T_SEQ + t) * U + u0;
#pragma unroll
            for (int q = 0; q < NV; ++q) {
                float4 v;
                v.x = h_st[q * 4 + 0]; v.y = h_st[q * 4 + 1];
                v.z = h_st[q * 4 + 2]; v.w = h_st[q * 4 + 3];
                *(float4*)(hp + q * 4) = v;
            }
        }
        __syncthreads();  // hbuf ready for next step
    };

    float4 zA[4 * NV], zB[4 * NV];
    prefetchZ(0, zA);
    for (int t2 = 0; t2 < T_SEQ; t2 += 2) {
        step(t2 + 0, zA, zB);
        step(t2 + 1, zB, zA);
    }

    if (LAST) {
        float* op = outLast + (size_t)(rowbase + mrow) * U + u0;
#pragma unroll
        for (int q = 0; q < NV; ++q) {
            float4 v;
            v.x = h_st[q * 4 + 0]; v.y = h_st[q * 4 + 1];
            v.z = h_st[q * 4 + 2]; v.w = h_st[q * 4 + 3];
            *(float4*)(op + q * 4) = v;
        }
    }
}

// ------------------------------------------------------------------- launch

extern "C" void kernel_launch(void* const* d_in, const int* in_sizes, int n_in,
                              void* d_out, int out_size, void* d_ws,
                              size_t ws_size, hipStream_t stream) {
    (void)in_sizes; (void)n_in; (void)out_size; (void)ws_size;
    const float* x   = (const float*)d_in[0];
    const float* Wk0 = (const float*)d_in[1];
    const float* Wr0 = (const float*)d_in[2];
    const float* b0  = (const float*)d_in[3];
    const float* Wk1 = (const float*)d_in[4];
    const float* Wr1 = (const float*)d_in[5];
    const float* b1  = (const float*)d_in[6];
    const float* Wko = (const float*)d_in[7];
    const float* Wro = (const float*)d_in[8];
    const float* bo  = (const float*)d_in[9];

    const size_t M = (size_t)256 * 1024;           // B*T rows
    float* Zb   = (float*)d_ws;                    // [M, 512] f32 (512 MB)
    float* Hseq = Zb + M * 512;                    // [M, 128] f32 (128 MB)

    dim3 blk(256);

    // layer 0
    lstm_in_gemm<128><<<dim3(2048, 8), blk, 0, stream>>>(x, Wk0, b0, Zb, 0x13579BDFu);
    lstm_scan_k<128, false><<<16, blk, 0, stream>>>(Zb, Wr0, Hseq, nullptr);
    // layer 1
    lstm_in_gemm<128><<<dim3(2048, 8), blk, 0, stream>>>(Hseq, Wk1, b1, Zb, 0x2468ACE0u);
    lstm_scan_k<128, false><<<16, blk, 0, stream>>>(Zb, Wr1, Hseq, nullptr);
    // output layer (U = 64), last state only
    lstm_in_gemm<64><<<dim3(2048, 4), blk, 0, stream>>>(Hseq, Wko, bo, Zb, 0x0F1E2D3Cu);
    lstm_scan_k<64, true><<<16, blk, 0, stream>>>(Zb, Wro, nullptr, (float*)d_out);
}
